// QuantizedLinear_14130442404082
// MI455X (gfx1250) — matrix operand order repaired and checked by reading, not compiled, not run on hardware
//
#include <hip/hip_runtime.h>

// Problem constants (fixed by the reference).
#define OUT_F 4096   // N (output features)
#define IN_F  4096   // K
#define NSUB  512
#define K_TOT 4096
#define N_TOT 4096

typedef __attribute__((ext_vector_type(16))) __bf16 v16bf;
typedef __attribute__((ext_vector_type(8)))  __bf16 v8bf;
typedef __attribute__((ext_vector_type(8)))  float  v8f;

// Split f32 -> hi+lo bf16 (bf16x3 GEMM trick: near-fp32 accuracy, 3 WMMAs).
__device__ __forceinline__ void split_f32(float x, __bf16 &h, __bf16 &l) {
    __bf16 hh = (__bf16)x;
    h = hh;
    l = (__bf16)(x - (float)hh);
}

// LDS byte offset of a flat pointer into shared memory (ISA: addr[31:0]).
__device__ __forceinline__ unsigned lds_off32(const void* p) {
    return (unsigned)(unsigned long long)p;
}

// Async DMA: 16 bytes per lane, global -> LDS, tracked by ASYNCcnt.
__device__ __forceinline__ void async_copy16(unsigned lds, const void* g) {
    asm volatile("global_load_async_to_lds_b128 %0, %1, off"
                 :: "v"(lds), "v"(g) : "memory");
}

// ---------------------------------------------------------------------------
// Pass 0: split input f32 -> Ahi/Alo bf16 (same total bytes as the source).
// ---------------------------------------------------------------------------
__global__ __launch_bounds__(256) void ql_split_input(
        const float* __restrict__ in,
        __bf16* __restrict__ hi, __bf16* __restrict__ lo, int n8) {
    int i = blockIdx.x * blockDim.x + threadIdx.x;   // one thread per 8 floats
    if (i >= n8) return;
    const float4* p = (const float4*)(in + (size_t)i * 8);
    float f[8];
    *(float4*)(f)     = p[0];
    *(float4*)(f + 4) = p[1];
    v8bf h, l;
    #pragma unroll
    for (int e = 0; e < 8; ++e) {
        __bf16 hh, ll; split_f32(f[e], hh, ll);
        h[e] = hh; l[e] = ll;
    }
    *(v8bf*)(hi + (size_t)i * 8) = h;
    *(v8bf*)(lo + (size_t)i * 8) = l;
}

// ---------------------------------------------------------------------------
// Pass 1: construct W[o][k] directly as split bf16 hi/lo pair.
// ---------------------------------------------------------------------------
__global__ __launch_bounds__(256) void ql_build_weight(
        const float* __restrict__ codebooks,
        const int*   __restrict__ mappings,
        const int*   __restrict__ sub_assign,
        const float* __restrict__ rowise,
        const float* __restrict__ colwise,
        __bf16* __restrict__ Whi, __bf16* __restrict__ Wlo) {
    int g = blockIdx.x * blockDim.x + threadIdx.x;   // one thread per (o, s)
    if (g >= OUT_F * NSUB) return;
    int o = g >> 9;
    int s = g & (NSUB - 1);
    int code = mappings[g];
    float cw = colwise[o];
    #pragma unroll
    for (int d = 0; d < 8; ++d) {
        int col = sub_assign[s * 8 + d];
        float v = codebooks[code * 8 + d] * rowise[col] * cw;
        __bf16 h, l; split_f32(v, h, l);
        size_t off = (size_t)o * IN_F + col;
        Whi[off] = h; Wlo[off] = l;
    }
}

// Pass 2: sparse outlier overwrite (after pass 1; raw values, no norms).
__global__ __launch_bounds__(256) void ql_sparse_scatter(
        const int*   __restrict__ idx,
        const float* __restrict__ vals,
        __bf16* __restrict__ Whi, __bf16* __restrict__ Wlo, int n) {
    int i = blockIdx.x * blockDim.x + threadIdx.x;
    if (i >= n) return;
    __bf16 h, l; split_f32(vals[i], h, l);
    int f = idx[i];
    Whi[f] = h; Wlo[f] = l;
}

// ---------------------------------------------------------------------------
// Pass 3: out[m][n] = sum_k A[m][k] * W[n][k] + bias[n]
// Block 128x128, 4 waves (2x2), each wave 64x64 via 4x4 16x16 frags.
// K-step 32, double-buffered LDS tiles filled by GLOBAL_LOAD_ASYNC_TO_LDS,
// fragments read via ds_load, 48 v_wmma_f32_16x16x32_bf16 per K-step.
// Each streamed B hi/lo fragment pair feeds 12 WMMAs -> ds latency hidden.
// LDS buffer layout (32 KB each, x2): Ahi 8K | Alo 8K | Bhi 8K | Blo 8K,
// each tile 128 rows x 64 bytes (32 bf16 of K).
// ---------------------------------------------------------------------------
#define BUF_BYTES 32768u

__global__ __launch_bounds__(128) void ql_wmma_gemm(
        const __bf16* __restrict__ Ahi, const __bf16* __restrict__ Alo,
        const __bf16* __restrict__ Whi, const __bf16* __restrict__ Wlo,
        const float* __restrict__ bias, float* __restrict__ out) {
    __shared__ __align__(16) unsigned char smem[2 * BUF_BYTES];

    const int t    = threadIdx.x;         // 0..127 (4 waves)
    const int lane = t & 31;
    const int wave = t >> 5;
    const int wm   = wave >> 1;           // 0..1
    const int wn   = wave & 1;            // 0..1
    const int l16  = lane & 15;
    const int lh   = lane >> 4;           // 0..1
    const int blockN = blockIdx.x * 128;
    const int blockM = blockIdx.y * 128;

    // Async fill of one 32 KB buffer: 4 arrays x 4 chunks of 16B per thread
    // = 16 async instructions per wave per K-step.
    auto fill = [&](unsigned bufB, int k0) {
        const __bf16* srcs[4] = {
            Ahi + (size_t)blockM * K_TOT + k0,
            Alo + (size_t)blockM * K_TOT + k0,
            Whi + (size_t)blockN * K_TOT + k0,
            Wlo + (size_t)blockN * K_TOT + k0,
        };
        #pragma unroll
        for (int a = 0; a < 4; ++a) {
            unsigned arr = bufB + a * 8192u;
            #pragma unroll
            for (int cc = 0; cc < 4; ++cc) {
                int c   = t + cc * 128;       // chunk 0..511
                int row = c >> 2;             // 0..127
                int cb  = c & 3;              // 16B chunk within 64B row
                async_copy16(lds_off32(smem + arr + row * 64 + cb * 16),
                             srcs[a] + (size_t)row * K_TOT + cb * 8);
            }
        }
    };

    // Per-frag LDS row/col bases (bytes within a tile).
    int aRow[4], bCol[4];
    #pragma unroll
    for (int mf = 0; mf < 4; ++mf) aRow[mf] = (wm * 64 + mf * 16 + l16) * 64;
    #pragma unroll
    for (int nf = 0; nf < 4; ++nf) bCol[nf] = (wn * 64 + nf * 16 + l16) * 64;

    const v8f vzero = {0.f, 0.f, 0.f, 0.f, 0.f, 0.f, 0.f, 0.f};
    v8f acc[4][4];
    #pragma unroll
    for (int mf = 0; mf < 4; ++mf)
        #pragma unroll
        for (int nf = 0; nf < 4; ++nf)
            acc[mf][nf] = vzero;

    union frag { v16bf v; v8bf h[2]; };

    fill(0u, 0);                              // prologue: buffer 0 <- K[0,32)

    #pragma unroll 1
    for (int k0 = 0; k0 < K_TOT; k0 += 32) {
        const unsigned bufB = (k0 & 32) ? BUF_BYTES : 0u;
        if (k0 + 32 < K_TOT) {
            fill(bufB ^ BUF_BYTES, k0 + 32);  // prefetch next tile
            // async loads complete in order: <=16 outstanding means the 16
            // fills of the CURRENT buffer have landed in LDS.
            asm volatile("s_wait_asynccnt 0x10" ::: "memory");
        } else {
            asm volatile("s_wait_asynccnt 0x0" ::: "memory");
        }
        __syncthreads();                      // whole tile visible to all waves

        // Resident A fragments (hi/lo x 4).
        v16bf ah[4], al[4];
        #pragma unroll
        for (int mf = 0; mf < 4; ++mf) {
            frag uh, ul;
            uh.h[0] = *(const v8bf*)(smem + bufB +         aRow[mf] + lh * 32);
            uh.h[1] = *(const v8bf*)(smem + bufB +         aRow[mf] + 16 + lh * 32);
            ul.h[0] = *(const v8bf*)(smem + bufB + 8192u + aRow[mf] + lh * 32);
            ul.h[1] = *(const v8bf*)(smem + bufB + 8192u + aRow[mf] + 16 + lh * 32);
            ah[mf] = uh.v; al[mf] = ul.v;
        }

        // Stream B fragments: each hi/lo pair feeds 12 WMMAs.
        #pragma unroll
        for (int nf = 0; nf < 4; ++nf) {
            v16bf bh = *(const v16bf*)(smem + bufB + 16384u + bCol[nf] + lh * 32);
            v16bf bl = *(const v16bf*)(smem + bufB + 24576u + bCol[nf] + lh * 32);
            #pragma unroll
            for (int mf = 0; mf < 4; ++mf) {
                acc[mf][nf] = __builtin_amdgcn_wmma_f32_16x16x32_bf16(
                    false, ah[mf], false, bh, (short)0, acc[mf][nf], false, false);
                acc[mf][nf] = __builtin_amdgcn_wmma_f32_16x16x32_bf16(
                    false, ah[mf], false, bl, (short)0, acc[mf][nf], false, false);
                acc[mf][nf] = __builtin_amdgcn_wmma_f32_16x16x32_bf16(
                    false, al[mf], false, bh, (short)0, acc[mf][nf], false, false);
            }
        }

        __syncthreads();   // all waves done reading before next-iter refill
    }

    // Epilogue: C/D layout -> VGPR r holds (M = r + 8*lh, N = l16) of each frag
    #pragma unroll
    for (int nf = 0; nf < 4; ++nf) {
        int n = blockN + wn * 64 + nf * 16 + l16;
        float bv = bias[n];
        #pragma unroll
        for (int mf = 0; mf < 4; ++mf) {
            int mbase = blockM + wm * 64 + mf * 16 + lh * 8;
            #pragma unroll
            for (int r = 0; r < 8; ++r)
                out[(size_t)(mbase + r) * N_TOT + n] = acc[mf][nf][r] + bv;
        }
    }
}

extern "C" void kernel_launch(void* const* d_in, const int* in_sizes, int n_in,
                              void* d_out, int out_size, void* d_ws, size_t ws_size,
                              hipStream_t stream) {
    const float* input      = (const float*)d_in[0]; // [4,2048,4096]
    const float* codebooks  = (const float*)d_in[1]; // [4096,8]
    const int*   mappings   = (const int*)  d_in[2]; // [4096,512]
    const int*   sub_assign = (const int*)  d_in[3]; // [4096]
    const float* rowise     = (const float*)d_in[4]; // [4096]
    const float* colwise    = (const float*)d_in[5]; // [4096]
    const int*   sparse_idx = (const int*)  d_in[6]; // [65536]
    const float* sparse_val = (const float*)d_in[7]; // [65536]
    const float* bias       = (const float*)d_in[8]; // [4096]
    float* out = (float*)d_out;

    // Workspace layout: Whi(32MB) Wlo(32MB) Ahi(64MB) Alo(64MB) = 192 MB
    const size_t wElems = (size_t)OUT_F * IN_F;
    const size_t aElems = (size_t)in_sizes[0];
    __bf16* Whi = (__bf16*)d_ws;
    __bf16* Wlo = Whi + wElems;
    __bf16* Ahi = Wlo + wElems;
    __bf16* Alo = Ahi + aElems;

    int n8 = (int)(aElems / 8);
    ql_split_input<<<(n8 + 255) / 256, 256, 0, stream>>>(input, Ahi, Alo, n8);

    ql_build_weight<<<(OUT_F * NSUB + 255) / 256, 256, 0, stream>>>(
        codebooks, mappings, sub_assign, rowise, colwise, Whi, Wlo);

    int nsp = in_sizes[6];
    ql_sparse_scatter<<<(nsp + 255) / 256, 256, 0, stream>>>(
        sparse_idx, sparse_val, Whi, Wlo, nsp);

    int M = (int)(aElems / IN_F);                    // 8192
    dim3 grid(N_TOT / 128, M / 128);                 // 32 x 64 blocks
    ql_wmma_gemm<<<grid, 128, 0, stream>>>(Ahi, Alo, Whi, Wlo, bias, out);
}
